// GAT_82910048682371
// MI455X (gfx1250) — matrix operand order repaired and checked by reading, not compiled, not run on hardware
//
#include <hip/hip_runtime.h>
#include <math.h>

typedef __attribute__((ext_vector_type(16))) __bf16        v16bf;
typedef __attribute__((ext_vector_type(8)))  float         v8f;
typedef __attribute__((ext_vector_type(4)))  float         v4f;
typedef __attribute__((ext_vector_type(4)))  unsigned int  v4u;
typedef __attribute__((ext_vector_type(2)))  unsigned int  v2u;

union Frag {
  v16bf v;
  v4u   q[2];
};

__device__ __forceinline__ unsigned short f2bf(float f) {
  unsigned int u = __float_as_uint(f);
  u = (u + 0x7FFFu + ((u >> 16) & 1u)) >> 16;   // round-to-nearest-even
  return (unsigned short)u;
}

__device__ __forceinline__ v8f wmma_bf16(v16bf a, v16bf b, v8f c) {
  // v_wmma_f32_16x16x32_bf16: D = A(16x32) * B(32x16) + C, f32 accum
  return __builtin_amdgcn_wmma_f32_16x16x32_bf16(false, a, false, b, (short)0, c,
                                                 false, false);
}

__device__ __forceinline__ v8f vzero8() {
  v8f z = {0.f, 0.f, 0.f, 0.f, 0.f, 0.f, 0.f, 0.f};
  return z;
}

// ---------------------------------------------------------------------------
// Kernel 0: pre-transpose / bf16-convert weights into column-major (N-major,
// K-contiguous) layouts so WMMA B-fragments are two contiguous b128 loads.
//   W0T[528][256] : cols 0..511 = w0_fc[h][k][d] (n = h*128+d), 512..515 =
//                   a0_self[h], 516..519 = a0_neigh[h], 520..527 = 0
//   W1T[528][512] : same for layer 1
//   FCT[256][512] : fc_w transposed
// ---------------------------------------------------------------------------
__global__ __launch_bounds__(256) void k_prep(
    const float* __restrict__ w0, const float* __restrict__ a0s, const float* __restrict__ a0n,
    const float* __restrict__ w1, const float* __restrict__ a1s, const float* __restrict__ a1n,
    const float* __restrict__ fcw,
    unsigned short* __restrict__ w0t, unsigned short* __restrict__ w1t,
    unsigned short* __restrict__ fct)
{
  const int i = blockIdx.x * 256 + threadIdx.x;
  const int NW0 = 528 * 256;
  const int NW1 = 528 * 512;
  const int NFC = 256 * 512;
  if (i < NW0) {
    const int n = i >> 8, k = i & 255;
    float v = 0.f;
    if (n < 512)      v = w0[((size_t)(n >> 7) * 256 + k) * 128 + (n & 127)];
    else if (n < 516) v = a0s[(size_t)(n - 512) * 256 + k];
    else if (n < 520) v = a0n[(size_t)(n - 516) * 256 + k];
    w0t[(size_t)n * 256 + k] = f2bf(v);
  } else if (i < NW0 + NW1) {
    const int j = i - NW0;
    const int n = j >> 9, k = j & 511;
    float v = 0.f;
    if (n < 512)      v = w1[((size_t)(n >> 7) * 512 + k) * 128 + (n & 127)];
    else if (n < 516) v = a1s[(size_t)(n - 512) * 512 + k];
    else if (n < 520) v = a1n[(size_t)(n - 516) * 512 + k];
    w1t[(size_t)n * 512 + k] = f2bf(v);
  } else if (i < NW0 + NW1 + NFC) {
    const int l = i - NW0 - NW1;
    const int n = l >> 9, k = l & 511;
    fct[(size_t)n * 512 + k] = f2bf(fcw[(size_t)k * 256 + n]);
  }
}

// ---------------------------------------------------------------------------
// Kernel 1: fused layer-0 GAT for one batch element per workgroup.
// Rows: 0 = x0, 1..10 = x1, 11..260 = x2 (padded to 272). K=256.
// Pass 1: WMMA on the 16 augmented logit columns -> softmax attention weights.
// Pass 2: WMMA over 32 value col-tiles x 17 row-tiles; accumulators are
//         attention-weighted straight into LDS h0/h1 via ds_add_f32.
// ---------------------------------------------------------------------------
#define L0_A_OFF     0
#define L0_LGT_OFF   143616   // 272*264*2
#define L0_WJ1_OFF   152320   // + 272*8*4
#define L0_WJ0_OFF   156672   // + 272*4*4
#define L0_H1_OFF    156928   // + 16*4*4
#define L0_H0_OFF    177408   // + 10*512*4
#define L0_LDS_BYTES 179456   // + 512*4

__global__ __launch_bounds__(512) void k0_gat_layer0(
    const float* __restrict__ x0, const float* __restrict__ x1, const float* __restrict__ x2,
    const unsigned short* __restrict__ w0t,
    float* __restrict__ h0ws, float* __restrict__ h1ws)
{
  extern __shared__ unsigned char smem[];
  unsigned short* A     = (unsigned short*)(smem + L0_A_OFF);    // 272 x 264 bf16
  float*          lgt   = (float*)(smem + L0_LGT_OFF);           // 272 x 8
  float*          wJ1   = (float*)(smem + L0_WJ1_OFF);           // 272 x 4
  float*          wJ0   = (float*)(smem + L0_WJ0_OFF);           // 16 x 4
  float*          h1acc = (float*)(smem + L0_H1_OFF);            // 10 x 512
  float*          h0acc = (float*)(smem + L0_H0_OFF);            // 512

  const int b    = blockIdx.x;
  const int tid  = threadIdx.x;
  const int lane = tid & 31;
  const int wv   = tid >> 5;
  const int l15  = lane & 15;
  const int kl   = lane >> 4;

  for (int i = tid; i < 10 * 512; i += 512) h1acc[i] = 0.f;
  h0acc[tid & 511] = 0.f;
  for (int i = tid; i < 272 * 4; i += 512) wJ1[i] = 0.f;
  if (tid < 64) wJ0[tid] = 0.f;

  // ---- stage A: 261 rows x 256 f32 -> bf16 LDS (pitch 264 halves) ----
  for (int idx = tid; idx < 261 * 64; idx += 512) {
    const int row = idx >> 6;
    const int kq  = idx & 63;
    const float* src;
    if (row == 0)       src = x0 + (size_t)b * 256;
    else if (row <= 10) src = x1 + ((size_t)b * 10 + (row - 1)) * 256;
    else                src = x2 + ((size_t)b * 250 + (row - 11)) * 256;
    v4f f = *(const v4f*)(src + kq * 4);
    v2u u;
    u.x = (unsigned)f2bf(f.x) | ((unsigned)f2bf(f.y) << 16);
    u.y = (unsigned)f2bf(f.z) | ((unsigned)f2bf(f.w) << 16);
    *(v2u*)(A + row * 264 + kq * 4) = u;
  }
  for (int idx = tid; idx < 11 * 64; idx += 512) {   // zero pad rows 261..271
    const int row = 261 + (idx >> 6);
    const int kq  = idx & 63;
    v2u z = {0u, 0u};
    *(v2u*)(A + row * 264 + kq * 4) = z;
  }
  __syncthreads();

  // ---- pass 1: logit columns (W0T rows 512..527) ----
  for (int rt = wv; rt < 17; rt += 16) {
    v8f acc = vzero8();
    const unsigned short* ap = A + (rt * 16 + l15) * 264 + kl * 8;
    const unsigned short* bp = w0t + (size_t)(512 + l15) * 256 + kl * 8;
#pragma unroll
    for (int ks = 0; ks < 8; ++ks) {
      Frag a, bb;
      a.q[0]  = *(const v4u*)(ap + ks * 32);
      a.q[1]  = *(const v4u*)(ap + ks * 32 + 16);
      bb.q[0] = *(const v4u*)(bp + ks * 32);
      bb.q[1] = *(const v4u*)(bp + ks * 32 + 16);
      acc = wmma_bf16(a.v, bb.v, acc);
    }
    if (l15 < 8) {
      const int rbase = rt * 16 + (kl << 3);
#pragma unroll
      for (int r = 0; r < 8; ++r) lgt[(rbase + r) * 8 + l15] = acc[r];
    }
  }
  __syncthreads();

  // ---- attention softmax: wave g handles group g (g=0: j0 root, 1..10: j1) ----
  if (wv <= 10 && lane < 4) {
    const int h       = lane;
    const int g       = wv;
    const int S       = (g == 0) ? 10 : 25;
    const int selfRow = (g == 0) ? 0 : g;
    const float lself = lgt[selfRow * 8 + h];
    float zmax = -3.4e38f;
    for (int s = 0; s <= S; ++s) {
      const int row = (g == 0) ? s : ((s == 0) ? g : (10 + (g - 1) * 25 + s));
      float z = lself + lgt[row * 8 + 4 + h];
      z = (z < 0.f) ? 0.2f * z : z;
      zmax = fmaxf(zmax, z);
    }
    float ssum = 0.f;
    for (int s = 0; s <= S; ++s) {
      const int row = (g == 0) ? s : ((s == 0) ? g : (10 + (g - 1) * 25 + s));
      float z = lself + lgt[row * 8 + 4 + h];
      z = (z < 0.f) ? 0.2f * z : z;
      const float e = expf(z - zmax);
      ssum += e;
      if (g == 0) wJ0[row * 4 + h] = e; else wJ1[row * 4 + h] = e;
    }
    const float inv = 1.f / ssum;
    for (int s = 0; s <= S; ++s) {
      const int row = (g == 0) ? s : ((s == 0) ? g : (10 + (g - 1) * 25 + s));
      if (g == 0) wJ0[row * 4 + h] *= inv; else wJ1[row * 4 + h] *= inv;
    }
  }
  __syncthreads();

  // ---- pass 2: value GEMM + fused attention aggregation ----
  for (int ct = wv; ct < 32; ct += 16) {
    Frag bw[8];
    const int col = ct * 16 + l15;
    const unsigned short* bp = w0t + (size_t)col * 256 + kl * 8;
#pragma unroll
    for (int ks = 0; ks < 8; ++ks) {     // keep all B K-fragments in VGPRs
      bw[ks].q[0] = *(const v4u*)(bp + ks * 32);
      bw[ks].q[1] = *(const v4u*)(bp + ks * 32 + 16);
    }
    const int h = col >> 7;
    for (int rt = 0; rt < 17; ++rt) {
      v8f acc = vzero8();
      const unsigned short* ap = A + (rt * 16 + l15) * 264 + kl * 8;
#pragma unroll
      for (int ks = 0; ks < 8; ++ks) {
        Frag a;
        a.q[0] = *(const v4u*)(ap + ks * 32);
        a.q[1] = *(const v4u*)(ap + ks * 32 + 16);
        acc = wmma_bf16(a.v, bw[ks].v, acc);
      }
      const int rbase = rt * 16 + (kl << 3);
#pragma unroll
      for (int r = 0; r < 8; ++r) {
        const int row = rbase + r;
        if (row >= 261) break;
        const float v = acc[r];
        if (row >= 1) {
          const int p1 = (row <= 10) ? (row - 1) : ((row - 11) / 25);
          atomicAdd(&h1acc[p1 * 512 + col], wJ1[row * 4 + h] * v);
        }
        if (row <= 10) {
          atomicAdd(&h0acc[col], wJ0[row * 4 + h] * v);
        }
      }
    }
  }
  __syncthreads();

  for (int i = tid; i < 10 * 512; i += 512) h1ws[(size_t)b * 5120 + i] = h1acc[i];
  h0ws[(size_t)b * 512 + (tid & 511)] = h0acc[tid & 511];
}

// ---------------------------------------------------------------------------
// Kernel 2: fused layer-1 GAT (rows 0=h0, 1..10=h1, K=512) -> h0' per batch.
// ---------------------------------------------------------------------------
__global__ __launch_bounds__(256) void k1_gat_layer1(
    const unsigned short* __restrict__ w1t,
    const float* __restrict__ h0ws, const float* __restrict__ h1ws,
    float* __restrict__ h0p)
{
  __shared__ unsigned short A[16 * 528];   // 16 rows x K=512 (+16 pad) bf16
  __shared__ float lgt[16 * 8];
  __shared__ float wAt[16 * 4];
  __shared__ float hacc[512];

  const int b    = blockIdx.x;
  const int tid  = threadIdx.x;
  const int lane = tid & 31;
  const int wv   = tid >> 5;
  const int l15  = lane & 15;
  const int kl   = lane >> 4;

  for (int i = tid; i < 512; i += 256) hacc[i] = 0.f;

  for (int idx = tid; idx < 16 * 128; idx += 256) {
    const int row = idx >> 7;
    const int kq  = idx & 127;
    v2u u = {0u, 0u};
    if (row <= 10) {
      const float* src = (row == 0) ? (h0ws + (size_t)b * 512)
                                    : (h1ws + (size_t)b * 5120 + (size_t)(row - 1) * 512);
      v4f f = *(const v4f*)(src + kq * 4);
      u.x = (unsigned)f2bf(f.x) | ((unsigned)f2bf(f.y) << 16);
      u.y = (unsigned)f2bf(f.z) | ((unsigned)f2bf(f.w) << 16);
    }
    *(v2u*)(A + row * 528 + kq * 4) = u;
  }
  __syncthreads();

  if (wv == 0) {                       // logit tile
    v8f acc = vzero8();
    const unsigned short* ap = A + l15 * 528 + kl * 8;
    const unsigned short* bp = w1t + (size_t)(512 + l15) * 512 + kl * 8;
#pragma unroll
    for (int ks = 0; ks < 16; ++ks) {
      Frag a, bb;
      a.q[0]  = *(const v4u*)(ap + ks * 32);
      a.q[1]  = *(const v4u*)(ap + ks * 32 + 16);
      bb.q[0] = *(const v4u*)(bp + ks * 32);
      bb.q[1] = *(const v4u*)(bp + ks * 32 + 16);
      acc = wmma_bf16(a.v, bb.v, acc);
    }
    if (l15 < 8) {
      const int rbase = kl << 3;
#pragma unroll
      for (int r = 0; r < 8; ++r) lgt[(rbase + r) * 8 + l15] = acc[r];
    }
  }
  __syncthreads();

  if (wv == 0 && lane < 4) {
    const int h = lane;
    const float lself = lgt[h];
    float zmax = -3.4e38f;
    for (int s = 0; s <= 10; ++s) {
      float z = lself + lgt[s * 8 + 4 + h];
      z = (z < 0.f) ? 0.2f * z : z;
      zmax = fmaxf(zmax, z);
    }
    float ssum = 0.f;
    for (int s = 0; s <= 10; ++s) {
      float z = lself + lgt[s * 8 + 4 + h];
      z = (z < 0.f) ? 0.2f * z : z;
      const float e = expf(z - zmax);
      ssum += e;
      wAt[s * 4 + h] = e;
    }
    const float inv = 1.f / ssum;
    for (int s = 0; s <= 10; ++s) wAt[s * 4 + h] *= inv;
  }
  __syncthreads();

  for (int ct = wv; ct < 32; ct += 8) {
    const int col = ct * 16 + l15;
    const unsigned short* bp = w1t + (size_t)col * 512 + kl * 8;
    const unsigned short* ap = A + l15 * 528 + kl * 8;
    v8f acc = vzero8();
#pragma unroll
    for (int ks = 0; ks < 16; ++ks) {
      Frag a, bb;
      a.q[0]  = *(const v4u*)(ap + ks * 32);
      a.q[1]  = *(const v4u*)(ap + ks * 32 + 16);
      bb.q[0] = *(const v4u*)(bp + ks * 32);
      bb.q[1] = *(const v4u*)(bp + ks * 32 + 16);
      acc = wmma_bf16(a.v, bb.v, acc);
    }
    const int h     = col >> 7;
    const int rbase = kl << 3;
#pragma unroll
    for (int r = 0; r < 8; ++r) {
      const int row = rbase + r;
      if (row >= 11) break;
      atomicAdd(&hacc[col], wAt[row * 4 + h] * acc[r]);
    }
  }
  __syncthreads();

  for (int i = tid; i < 512; i += 256) h0p[(size_t)b * 512 + i] = hacc[i];
}

// ---------------------------------------------------------------------------
// Kernel 3: out(1024x256) = h0'(1024x512) @ fc_w(512x256), bf16 WMMA.
// One wave per 16x16 output tile.
// ---------------------------------------------------------------------------
__global__ __launch_bounds__(256) void k2_final(
    const float* __restrict__ h0p, const unsigned short* __restrict__ fct,
    float* __restrict__ out)
{
  const int tid   = threadIdx.x;
  const int lane  = tid & 31;
  const int wv    = tid >> 5;
  const int l15   = lane & 15;
  const int kl    = lane >> 4;
  const int wtile = blockIdx.x * 8 + wv;   // 0..1023
  const int rt    = wtile >> 4;            // 0..63
  const int ct    = wtile & 15;            // 0..15

  const float*          arow = h0p + (size_t)(rt * 16 + l15) * 512 + kl * 8;
  const unsigned short* bp   = fct + (size_t)(ct * 16 + l15) * 512 + kl * 8;
  v8f acc = vzero8();
#pragma unroll
  for (int ks = 0; ks < 16; ++ks) {
    Frag a, bb;
    v4f f0 = *(const v4f*)(arow + ks * 32);
    v4f f1 = *(const v4f*)(arow + ks * 32 + 4);
    v4f f2 = *(const v4f*)(arow + ks * 32 + 16);
    v4f f3 = *(const v4f*)(arow + ks * 32 + 20);
    a.q[0].x = (unsigned)f2bf(f0.x) | ((unsigned)f2bf(f0.y) << 16);
    a.q[0].y = (unsigned)f2bf(f0.z) | ((unsigned)f2bf(f0.w) << 16);
    a.q[0].z = (unsigned)f2bf(f1.x) | ((unsigned)f2bf(f1.y) << 16);
    a.q[0].w = (unsigned)f2bf(f1.z) | ((unsigned)f2bf(f1.w) << 16);
    a.q[1].x = (unsigned)f2bf(f2.x) | ((unsigned)f2bf(f2.y) << 16);
    a.q[1].y = (unsigned)f2bf(f2.z) | ((unsigned)f2bf(f2.w) << 16);
    a.q[1].z = (unsigned)f2bf(f3.x) | ((unsigned)f2bf(f3.y) << 16);
    a.q[1].w = (unsigned)f2bf(f3.z) | ((unsigned)f2bf(f3.w) << 16);
    bb.q[0] = *(const v4u*)(bp + ks * 32);
    bb.q[1] = *(const v4u*)(bp + ks * 32 + 16);
    acc = wmma_bf16(a.v, bb.v, acc);
  }
  const int orow = rt * 16 + (kl << 3);
  const int ocol = ct * 16 + l15;
#pragma unroll
  for (int r = 0; r < 8; ++r)
    out[(size_t)(orow + r) * 256 + ocol] = acc[r];
}

// ---------------------------------------------------------------------------
// Workspace layout (bytes, 256-aligned):
//   0        W0T  bf16 528x256   (270336)
//   270336   W1T  bf16 528x512   (540672)
//   811008   FCT  bf16 256x512   (262144)
//   1073152  h0   f32 1024x512   (2097152)
//   3170304  h1   f32 1024x10x512 (20971520)
//   24141824 h0'  f32 1024x512   (2097152)   total ~26.2 MB
// ---------------------------------------------------------------------------
extern "C" void kernel_launch(void* const* d_in, const int* in_sizes, int n_in,
                              void* d_out, int out_size, void* d_ws, size_t ws_size,
                              hipStream_t stream) {
  const float* x0  = (const float*)d_in[0];
  const float* x1  = (const float*)d_in[1];
  const float* x2  = (const float*)d_in[2];
  const float* w0  = (const float*)d_in[3];
  const float* a0s = (const float*)d_in[4];
  const float* a0n = (const float*)d_in[5];
  const float* w1  = (const float*)d_in[6];
  const float* a1s = (const float*)d_in[7];
  const float* a1n = (const float*)d_in[8];
  const float* fcw = (const float*)d_in[9];

  unsigned char* ws   = (unsigned char*)d_ws;
  unsigned short* w0t = (unsigned short*)(ws + 0);
  unsigned short* w1t = (unsigned short*)(ws + 270336);
  unsigned short* fct = (unsigned short*)(ws + 811008);
  float* h0ws         = (float*)(ws + 1073152);
  float* h1ws         = (float*)(ws + 3170304);
  float* h0p          = (float*)(ws + 24141824);

  k_prep<<<2096, 256, 0, stream>>>(w0, a0s, a0n, w1, a1s, a1n, fcw, w0t, w1t, fct);
  k0_gat_layer0<<<1024, 512, L0_LDS_BYTES, stream>>>(x0, x1, x2, w0t, h0ws, h1ws);
  k1_gat_layer1<<<1024, 256, 0, stream>>>(w1t, h0ws, h1ws, h0p);
  k2_final<<<128, 256, 0, stream>>>(h0p, fct, (float*)d_out);
}
